// BertAttention_72748156060131
// MI455X (gfx1250) — matrix operand order repaired and checked by reading, not compiled, not run on hardware
//
#include <hip/hip_runtime.h>
#include <hip/hip_bf16.h>

// DeBERTa disentangled attention for MI455X (gfx1250, wave32).
// All GEMMs run on v_wmma_f32_16x16x32_bf16 (f32 accumulate).
// B=1, S=2048, H=768, NH=12, D=64, SPAN=512.

#define SEQ     2048
#define HID     768
#define NHEAD   12
#define HDIM    64
#define SPAN_   512
#define L2SPAN  1024
#define NEG_INF (-3.4028235e38f)

typedef __attribute__((ext_vector_type(16))) __bf16 v16bf;
typedef __attribute__((ext_vector_type(8)))  __bf16 v8bf;
typedef __attribute__((ext_vector_type(8)))  float  v8f;

__device__ __forceinline__ __bf16 f2bf(float f) {
  unsigned u = __builtin_bit_cast(unsigned, f);
  unsigned r = u + 0x7FFFu + ((u >> 16) & 1u);   // round-to-nearest-even
  unsigned short h = (unsigned short)(r >> 16);
  return __builtin_bit_cast(__bf16, h);
}

__device__ __forceinline__ int iclamp(int v, int lo, int hi) {
  int a = v > lo ? v : lo;          // v_max_i32
  return a < hi ? a : hi;           // v_min_i32
}

// ---- WMMA operand loads (16-bit, k contiguous per lane half, wave32) ----------------
// Offsets are 32-bit from a uniform base so the backend can use
// SADDR + 32-bit voffset global loads (no per-lane 64-bit address math).
// A 16x32 (MxK): lanes 0-15 row M=lane, K=k0..k0+15;
//                lanes 16-31 same rows, K=k0+16..k0+31.
__device__ __forceinline__ v16bf load_a16x32(const __bf16* __restrict__ A,
                                             unsigned off) {
  const __bf16* p = A + off;
  v8bf lo = *(const v8bf*)(p);
  v8bf hi = *(const v8bf*)(p + 8);
  v16bf v;
#pragma unroll
  for (int t = 0; t < 8; ++t) { v[t] = lo[t]; v[t + 8] = hi[t]; }
  return v;
}
// B 32x16 (KxN) from Bt stored N-major (Bt[n*ldb + k]): lanes 0-15 hold
// N=lane, K=k0..k0+15; lanes 16-31 hold N=lane-16, K=k0+16..k0+31.
__device__ __forceinline__ v16bf load_b32x16(const __bf16* __restrict__ Bt,
                                             unsigned off) {
  return *(const v16bf*)(Bt + off);
}

// ---- generic bf16 WMMA GEMM: C[z] = A[z] @ Bt[z]^T (+bias)(+resid), *scale -
// Wave tile 16(M) x 64(N): one A load feeds 4 v_wmma issues. 8 waves/block
// arranged (8/WN) x WN -> block tile (128/WN x WN*64).
template <int WN, bool OUT_BF16, bool CT>
__global__ __launch_bounds__(256) void gemm_bf16_kernel(
    const __bf16* __restrict__ A,  long sAz,
    const __bf16* __restrict__ Bt, long sBz,
    void* __restrict__ C,          long sCz,
    int K, int lda, int ldb, int ldc,
    const float* __restrict__ bias,  long sbz,
    const float* __restrict__ resid, long srz,
    float scale) {
  constexpr int WM = 8 / WN;
  const int z    = blockIdx.z;
  const int lane = threadIdx.x & 31;
  const int wave = threadIdx.x >> 5;
  const int i0 = blockIdx.y * (WM * 16) + (wave / WN) * 16;
  const int n0 = blockIdx.x * (WN * 64) + (wave % WN) * 64;
  const __bf16* Az  = A  + (long)z * sAz;     // uniform per-z bases
  const __bf16* Btz = Bt + (long)z * sBz;

  const unsigned aRow = (unsigned)(i0 + (lane & 15)) * (unsigned)lda
                      + ((lane >> 4) << 4);
  const unsigned bRow = (unsigned)(n0 + (lane & 15)) * (unsigned)ldb
                      + ((lane >> 4) << 4);

  v8f acc[4] = {{}, {}, {}, {}};
  for (int k0 = 0; k0 < K; k0 += 32) {
    v16bf a = load_a16x32(Az, aRow + (unsigned)k0);
#pragma unroll
    for (int t = 0; t < 4; ++t) {
      v16bf b = load_b32x16(Btz, bRow + (unsigned)(t * 16 * ldb) + (unsigned)k0);
      acc[t] = __builtin_amdgcn_wmma_f32_16x16x32_bf16(false, a, false, b,
                                                       (short)0, acc[t],
                                                       false, false);
    }
  }

  const int row_base = i0 + ((lane >> 4) << 3);
  const int col      = lane & 15;
  __bf16* Cb = (__bf16*)C + (long)z * sCz;
  float*  Cf = (float*)C + (long)z * sCz;
  const float* biasz  = bias  ? bias  + (long)z * sbz : nullptr;
  const float* residz = resid ? resid + (long)z * srz : nullptr;
#pragma unroll
  for (int t = 0; t < 4; ++t) {
    const int n = n0 + t * 16 + col;
    const float bn = biasz ? biasz[n] : 0.0f;
#pragma unroll
    for (int r = 0; r < 8; ++r) {
      const int m = row_base + r;
      float v = acc[t][r] * scale + bn;
      if (residz) v += residz[(unsigned)(m * ldc + n)];
      const unsigned off = CT ? (unsigned)(n * ldc + m)
                              : (unsigned)(m * ldc + n);
      if (OUT_BF16) Cb[off] = f2bf(v);
      else          Cf[off] = v;
    }
  }
}

// ---- scores = scale*Q@K^T + gather(c2p) + gather(p2c) + mask ---------------
// Wave tile 16x64 (4 accumulators); 8 waves as 4x2 -> block tile 64x128.
__global__ __launch_bounds__(256) void scores_kernel(
    const __bf16* __restrict__ Qh, const __bf16* __restrict__ Kh,
    const float* __restrict__ c2p, const float* __restrict__ p2c,
    const int* __restrict__ amask, float* __restrict__ Sc, float scale) {
  const int z    = blockIdx.z;
  const int lane = threadIdx.x & 31;
  const int wave = threadIdx.x >> 5;
  const int i0 = blockIdx.y * 64 + (wave >> 1) * 16;
  const int n0 = blockIdx.x * 128 + (wave & 1) * 64;
  const __bf16* A  = Qh + (long)z * (SEQ * HDIM);   // uniform per-z bases
  const __bf16* Bt = Kh + (long)z * (SEQ * HDIM);
  const float* c2ph = c2p + (long)z * (SEQ * L2SPAN);
  const float* p2ch = p2c + (long)z * (SEQ * L2SPAN);
  float* Sh = Sc + (long)z * ((long)SEQ * SEQ);

  const unsigned aRow = (unsigned)(i0 + (lane & 15)) * HDIM + ((lane >> 4) << 4);
  const unsigned bRow = (unsigned)(n0 + (lane & 15)) * HDIM + ((lane >> 4) << 4);

  v8f acc[4] = {{}, {}, {}, {}};
#pragma unroll
  for (int k0 = 0; k0 < HDIM; k0 += 32) {
    v16bf a = load_a16x32(A, aRow + (unsigned)k0);
#pragma unroll
    for (int t = 0; t < 4; ++t) {
      v16bf b = load_b32x16(Bt, bRow + (unsigned)(t * 16 * HDIM) + (unsigned)k0);
      acc[t] = __builtin_amdgcn_wmma_f32_16x16x32_bf16(false, a, false, b,
                                                       (short)0, acc[t],
                                                       false, false);
    }
  }

  const int row_base = i0 + ((lane >> 4) << 3);
  const int col      = lane & 15;
  // Hoist row masks (rows invariant across the 4 N-tiles) -> branchless select.
  int rmask[8];
#pragma unroll
  for (int r = 0; r < 8; ++r) rmask[r] = amask[row_base + r];

#pragma unroll
  for (int t = 0; t < 4; ++t) {
    const int j  = n0 + t * 16 + col;
    const int mj = amask[j];
    const unsigned jRow = (unsigned)j << 10;        // j * L2SPAN
#pragma unroll
    for (int r = 0; r < 8; ++r) {
      const int i = row_base + r;
      const unsigned d = (unsigned)iclamp(i - j + SPAN_, 0, L2SPAN - 1);
      float v = acc[t][r] * scale
              + c2ph[((unsigned)i << 10) + d]
              + p2ch[jRow + d];
      v = (rmask[r] && mj) ? v : NEG_INF;
      Sh[((unsigned)i << 11) + (unsigned)j] = v;    // i * SEQ + j
    }
  }
}

// ---- row softmax over 2048, f32 scores -> bf16 probs -----------------------
__global__ __launch_bounds__(256) void softmax_kernel(
    const float* __restrict__ Sc, __bf16* __restrict__ P) {
  const long base = ((long)blockIdx.y * SEQ + blockIdx.x) * SEQ;
  __shared__ float red[256];
  const int t = threadIdx.x;
  float mx = NEG_INF;
  for (int j = t; j < SEQ; j += 256) mx = fmaxf(mx, Sc[base + j]);
  red[t] = mx; __syncthreads();
  for (int s = 128; s > 0; s >>= 1) {
    if (t < s) red[t] = fmaxf(red[t], red[t + s]);
    __syncthreads();
  }
  mx = red[0]; __syncthreads();
  float sum = 0.f;
  for (int j = t; j < SEQ; j += 256) sum += __expf(Sc[base + j] - mx);
  red[t] = sum; __syncthreads();
  for (int s = 128; s > 0; s >>= 1) {
    if (t < s) red[t] += red[t + s];
    __syncthreads();
  }
  const float inv = 1.0f / red[0];
  for (int j = t; j < SEQ; j += 256)
    P[base + j] = f2bf(__expf(Sc[base + j] - mx) * inv);
}

// ---- LayerNorm over H=768 --------------------------------------------------
__global__ __launch_bounds__(256) void layernorm_kernel(
    const float* __restrict__ X, const float* __restrict__ gamma,
    const float* __restrict__ beta, float* __restrict__ Out) {
  const float* x = X + (long)blockIdx.x * HID;
  float* o = Out + (long)blockIdx.x * HID;
  __shared__ float red[256], red2[256];
  const int t = threadIdx.x;
  float s = 0.f, s2 = 0.f;
  for (int c = t; c < HID; c += 256) { float v = x[c]; s += v; s2 += v * v; }
  red[t] = s; red2[t] = s2; __syncthreads();
  for (int st = 128; st > 0; st >>= 1) {
    if (t < st) { red[t] += red[t + st]; red2[t] += red2[t + st]; }
    __syncthreads();
  }
  const float mu   = red[0] / HID;
  const float var  = red2[0] / HID - mu * mu;
  const float rstd = rsqrtf(var + 1e-7f);
  for (int c = t; c < HID; c += 256)
    o[c] = (x[c] - mu) * rstd * gamma[c] + beta[c];
}

// ---- f32 -> bf16 elementwise ----------------------------------------------
__global__ void cvt_bf16_kernel(const float* __restrict__ src,
                                __bf16* __restrict__ dst, long n) {
  long i = (long)blockIdx.x * blockDim.x + threadIdx.x;
  if (i < n) dst[i] = f2bf(src[i]);
}

// ---- f32 KxN -> bf16 NxK (LDS-tiled transpose) -----------------------------
__global__ void transpose_cvt_kernel(const float* __restrict__ src,
                                     __bf16* __restrict__ dst,
                                     int rows, int cols) {
  __shared__ float tile[32][33];
  const int c0 = blockIdx.x * 32, r0 = blockIdx.y * 32;
  const int tx = threadIdx.x, ty = threadIdx.y;  // 32x8
#pragma unroll
  for (int dy = 0; dy < 32; dy += 8)
    tile[ty + dy][tx] = src[(long)(r0 + ty + dy) * cols + (c0 + tx)];
  __syncthreads();
#pragma unroll
  for (int dy = 0; dy < 32; dy += 8)
    dst[(long)(c0 + ty + dy) * rows + (r0 + tx)] = f2bf(tile[tx][ty + dy]);
}

// ---------------------------------------------------------------------------
extern "C" void kernel_launch(void* const* d_in, const int* in_sizes, int n_in,
                              void* d_out, int out_size, void* d_ws, size_t ws_size,
                              hipStream_t stream) {
  const float* hs    = (const float*)d_in[0];
  const int*   amask = (const int*)d_in[1];
  const float* Wq  = (const float*)d_in[2];
  const float* bq  = (const float*)d_in[3];
  const float* Wk  = (const float*)d_in[4];
  const float* bk  = (const float*)d_in[5];
  const float* Wv  = (const float*)d_in[6];
  const float* bv  = (const float*)d_in[7];
  const float* rel = (const float*)d_in[8];
  const float* Wpk = (const float*)d_in[9];
  const float* Wpq = (const float*)d_in[10];
  const float* bpq = (const float*)d_in[11];
  const float* Wo  = (const float*)d_in[12];
  const float* bo  = (const float*)d_in[13];
  const float* gamma = (const float*)d_in[14];
  const float* beta  = (const float*)d_in[15];
  float* out = (float*)d_out;

  const float scale = 1.0f / sqrtf((float)(HDIM * 3));

  // workspace carve-up (256B aligned)
  size_t off = 0;
  auto wsalloc = [&](size_t bytes) -> void* {
    off = (off + 255) & ~(size_t)255;
    void* p = (char*)d_ws + off;
    off += bytes;
    return p;
  };
  __bf16* hsb  = (__bf16*)wsalloc((size_t)SEQ * HID * 2);
  __bf16* relb = (__bf16*)wsalloc((size_t)L2SPAN * HID * 2);
  __bf16* WqT  = (__bf16*)wsalloc((size_t)HID * HID * 2);
  __bf16* WkT  = (__bf16*)wsalloc((size_t)HID * HID * 2);
  __bf16* WvT  = (__bf16*)wsalloc((size_t)HID * HID * 2);
  __bf16* WoT  = (__bf16*)wsalloc((size_t)HID * HID * 2);
  __bf16* WpkT = (__bf16*)wsalloc((size_t)HID * HID * 2);
  __bf16* WpqT = (__bf16*)wsalloc((size_t)HID * HID * 2);
  __bf16* Qh   = (__bf16*)wsalloc((size_t)NHEAD * SEQ * HDIM * 2);
  __bf16* Kh   = (__bf16*)wsalloc((size_t)NHEAD * SEQ * HDIM * 2);
  __bf16* Vt   = (__bf16*)wsalloc((size_t)NHEAD * HDIM * SEQ * 2);  // head-transposed
  __bf16* PosK = (__bf16*)wsalloc((size_t)NHEAD * L2SPAN * HDIM * 2);
  __bf16* PosQ = (__bf16*)wsalloc((size_t)NHEAD * L2SPAN * HDIM * 2);
  float*  c2p  = (float*)wsalloc((size_t)NHEAD * SEQ * L2SPAN * 4);
  float*  p2c  = (float*)wsalloc((size_t)NHEAD * SEQ * L2SPAN * 4);
  float*  scor = (float*)wsalloc((size_t)NHEAD * SEQ * SEQ * 4);
  __bf16* prob = (__bf16*)wsalloc((size_t)NHEAD * SEQ * SEQ * 2);
  __bf16* ctxb = (__bf16*)wsalloc((size_t)SEQ * HID * 2);
  float*  aout = (float*)wsalloc((size_t)SEQ * HID * 4);

  // 1) bf16 conversions / weight transposes
  {
    long n1 = (long)SEQ * HID, n2 = (long)L2SPAN * HID;
    cvt_bf16_kernel<<<dim3((n1 + 255) / 256), 256, 0, stream>>>(hs, hsb, n1);
    cvt_bf16_kernel<<<dim3((n2 + 255) / 256), 256, 0, stream>>>(rel, relb, n2);
    dim3 tg(HID / 32, HID / 32), tb(32, 8);
    transpose_cvt_kernel<<<tg, tb, 0, stream>>>(Wq,  WqT,  HID, HID);
    transpose_cvt_kernel<<<tg, tb, 0, stream>>>(Wk,  WkT,  HID, HID);
    transpose_cvt_kernel<<<tg, tb, 0, stream>>>(Wv,  WvT,  HID, HID);
    transpose_cvt_kernel<<<tg, tb, 0, stream>>>(Wo,  WoT,  HID, HID);
    transpose_cvt_kernel<<<tg, tb, 0, stream>>>(Wpk, WpkT, HID, HID);
    transpose_cvt_kernel<<<tg, tb, 0, stream>>>(Wpq, WpqT, HID, HID);
  }

  // 2) projections (grid.z = head, N=64 per head -> WN=1, 128x64 blocks)
  {
    dim3 g(1, SEQ / 128, NHEAD);
    gemm_bf16_kernel<1, true, false><<<g, 256, 0, stream>>>(
        hsb, 0L, WqT, (long)HDIM * HID, Qh, (long)SEQ * HDIM,
        HID, HID, HID, HDIM, bq, HDIM, nullptr, 0L, 1.0f);
    gemm_bf16_kernel<1, true, false><<<g, 256, 0, stream>>>(
        hsb, 0L, WkT, (long)HDIM * HID, Kh, (long)SEQ * HDIM,
        HID, HID, HID, HDIM, bk, HDIM, nullptr, 0L, 1.0f);
    gemm_bf16_kernel<1, true, true><<<g, 256, 0, stream>>>(   // V stored D x S
        hsb, 0L, WvT, (long)HDIM * HID, Vt, (long)HDIM * SEQ,
        HID, HID, HID, SEQ, bv, HDIM, nullptr, 0L, 1.0f);
    dim3 gp(1, L2SPAN / 128, NHEAD);
    gemm_bf16_kernel<1, true, false><<<gp, 256, 0, stream>>>(
        relb, 0L, WpkT, (long)HDIM * HID, PosK, (long)L2SPAN * HDIM,
        HID, HID, HID, HDIM, nullptr, 0L, nullptr, 0L, 1.0f);
    gemm_bf16_kernel<1, true, false><<<gp, 256, 0, stream>>>(
        relb, 0L, WpqT, (long)HDIM * HID, PosQ, (long)L2SPAN * HDIM,
        HID, HID, HID, HDIM, bpq, HDIM, nullptr, 0L, 1.0f);
  }

  // 3) c2p = scale * Q @ PosK^T ; p2c = scale * K @ PosQ^T  (per head, WN=2)
  {
    dim3 g(L2SPAN / 128, SEQ / 64, NHEAD);
    gemm_bf16_kernel<2, false, false><<<g, 256, 0, stream>>>(
        Qh, (long)SEQ * HDIM, PosK, (long)L2SPAN * HDIM, c2p, (long)SEQ * L2SPAN,
        HDIM, HDIM, HDIM, L2SPAN, nullptr, 0L, nullptr, 0L, scale);
    gemm_bf16_kernel<2, false, false><<<g, 256, 0, stream>>>(
        Kh, (long)SEQ * HDIM, PosQ, (long)L2SPAN * HDIM, p2c, (long)SEQ * L2SPAN,
        HDIM, HDIM, HDIM, L2SPAN, nullptr, 0L, nullptr, 0L, scale);
  }

  // 4) scores = scale*QK^T + gathers + mask
  {
    dim3 g(SEQ / 128, SEQ / 64, NHEAD);
    scores_kernel<<<g, 256, 0, stream>>>(Qh, Kh, c2p, p2c, amask, scor, scale);
  }

  // 5) softmax -> bf16 probs
  softmax_kernel<<<dim3(SEQ, NHEAD), 256, 0, stream>>>(scor, prob);

  // 6) ctx = probs @ V  (per head, recombined into S x H row-major bf16)
  {
    dim3 g(1, SEQ / 128, NHEAD);
    gemm_bf16_kernel<1, true, false><<<g, 256, 0, stream>>>(
        prob, (long)SEQ * SEQ, Vt, (long)HDIM * SEQ, ctxb, (long)HDIM,
        SEQ, SEQ, SEQ, HID, nullptr, 0L, nullptr, 0L, 1.0f);
  }

  // 7) out = ctx @ Wo + bo + hidden_states  (f32)
  {
    dim3 g(1, SEQ / 128, NHEAD);
    gemm_bf16_kernel<1, false, false><<<g, 256, 0, stream>>>(
        ctxb, 0L, WoT, (long)HDIM * HID, aout, (long)HDIM,
        HID, HID, HID, HID, bo, HDIM, hs, HDIM, 1.0f);
  }

  // 8) LayerNorm -> d_out
  layernorm_kernel<<<dim3(SEQ), 256, 0, stream>>>(aout, gamma, beta, out);
}
